// HeteroGNN_18356690223631
// MI455X (gfx1250) — matrix operand order repaired and checked by reading, not compiled, not run on hardware
//
#include <hip/hip_runtime.h>
#include <hip/hip_bf16.h>
#include <math.h>

// ---------------- problem constants (match reference) ----------------
#define NNODE 40000
#define HID   512
#define NH    8
#define HD    64
#define NOUT  256

typedef __attribute__((ext_vector_type(16))) _Float16 v16h;
typedef __attribute__((ext_vector_type(2)))  _Float16 v2h;
typedef __attribute__((ext_vector_type(8)))  float    v8f;

// ---------------- WMMA tiled GEMM: C = A(MxK) @ B(KxN) (+bias) ----------------
// fp32 in/out, f16 multiply / f32 accumulate via v_wmma_f32_16x16x32_f16.
// * LDS tiles stored PRE-SWIZZLED in per-lane fragment order (32B vector ds loads)
// * double-buffered LDS, one barrier per K-step, staging overlaps WMMA
// * branch-free staging (row clamp) so global loads pipeline
// Batched via blockIdx.z (strides sA/sB/sC in elements).
#define BM 256
#define BN 64
#define BK 32
#define A_HALVES (16 * 32 * 16)   // 8192 halves per buffer
#define B_HALVES (4 * 32 * 16)    // 2048 halves per buffer

__global__ __launch_bounds__(256)
void wmma_gemm_kernel(const float* __restrict__ A, int lda, long sA,
                      const float* __restrict__ B, int ldb, long sB,
                      const float* __restrict__ bias,
                      float* __restrict__ C, int ldc, long sC,
                      int M, int Nn, int K)
{
    __shared__ __align__(32) _Float16 As[2][A_HALVES];   // 32 KB
    __shared__ __align__(32) _Float16 Bs[2][B_HALVES];   //  8 KB

    const int tid  = threadIdx.x;
    const int wave = tid >> 5;
    const int lane = tid & 31;
    const int half = lane >> 4;
    const int ln   = lane & 15;

    const long aoff = (long)blockIdx.z * sA;
    const long boff = (long)blockIdx.z * sB;
    const long coff = (long)blockIdx.z * sC;

    const int bm = blockIdx.x * BM;
    const int bn = blockIdx.y * BN;

    v8f acc[2][4];
    #pragma unroll
    for (int m = 0; m < 2; ++m)
        #pragma unroll
        for (int i = 0; i < 4; ++i)
            #pragma unroll
            for (int j = 0; j < 8; ++j) acc[m][i][j] = 0.0f;

    const int nsteps = K / BK;

    // stage K-tile s into LDS buffer s&1 (branch-free, load-then-store batches)
    auto stage = [&](int s) {
        const int kk = s * BK;
        const int pb = s & 1;
        // ---- A tile (256 x 32): 4096 pairs, two batches of 8 per thread ----
        #pragma unroll
        for (int hb = 0; hb < 2; ++hb) {
            float2 va[8];
            #pragma unroll
            for (int q = 0; q < 8; ++q) {
                int t = tid + (hb * 8 + q) * 256;
                int r = t >> 4, c = (t & 15) * 2;
                int row = bm + r; row = (row < M) ? row : (M - 1);   // clamp, no branch
                va[q] = *(const float2*)(A + aoff + (long)row * lda + kk + c);
            }
            #pragma unroll
            for (int q = 0; q < 8; ++q) {
                int t = tid + (hb * 8 + q) * 256;
                int r = t >> 4, c = (t & 15) * 2;
                // inverse of ISA 7.12.2 16-bit A layout
                int g = c >> 3, h2 = g & 1;
                int j = 2 * ((g >> 1) * 4 + ((c & 7) >> 1));
                int fl = h2 * 16 + (r & 15);
                *(v2h*)&As[pb][(((r >> 4) * 32 + fl) << 4) + j] =
                    (v2h){ (_Float16)va[q].x, (_Float16)va[q].y };
            }
        }
        // ---- B tile (32 x 64): 1024 pairs, 4 per thread ----
        float2 vb[4];
        #pragma unroll
        for (int q = 0; q < 4; ++q) {
            int t = tid + q * 256;
            int c = t & 63, r = (t >> 6) * 2;
            const float* bp = B + boff + (long)(kk + r) * ldb + bn + c;
            vb[q].x = bp[0];
            vb[q].y = bp[ldb];
        }
        #pragma unroll
        for (int q = 0; q < 4; ++q) {
            int t = tid + q * 256;
            int c = t & 63, r = (t >> 6) * 2;
            int h2 = (r >> 3) & 1;
            int j = 2 * ((r >> 4) * 4 + ((r & 7) >> 1));
            int fl = h2 * 16 + (c & 15);
            int nt = c >> 4;
            *(v2h*)&Bs[pb][((nt * 32 + fl) << 4) + j] =
                (v2h){ (_Float16)vb[q].x, (_Float16)vb[q].y };
        }
    };

    stage(0);
    for (int s = 0; s < nsteps; ++s) {
        __syncthreads();                         // buf s&1 ready; prior compute done
        if (s + 1 < nsteps) stage(s + 1);        // overlap next-tile staging with WMMA
        const int p = s & 1;
        v16h a0 = *(const v16h*)&As[p][(wave * 32 + lane) << 4];
        v16h a1 = *(const v16h*)&As[p][((8 + wave) * 32 + lane) << 4];
        #pragma unroll
        for (int nt = 0; nt < 4; ++nt) {
            v16h b = *(const v16h*)&Bs[p][(nt * 32 + lane) << 4];
            acc[0][nt] = __builtin_amdgcn_wmma_f32_16x16x32_f16(
                false, a0, false, b, (short)0, acc[0][nt], false, false);
            acc[1][nt] = __builtin_amdgcn_wmma_f32_16x16x32_f16(
                false, a1, false, b, (short)0, acc[1][nt], false, false);
        }
    }

    // epilogue: C layout (VGPR r -> M = r + 8*half, N = ln)
    #pragma unroll
    for (int ms = 0; ms < 2; ++ms) {
        #pragma unroll
        for (int nt = 0; nt < 4; ++nt) {
            #pragma unroll
            for (int r = 0; r < 8; ++r) {
                int row = bm + ms * 128 + wave * 16 + r + 8 * half;
                int col = bn + nt * 16 + ln;
                if (row < M) {
                    float v = acc[ms][nt][r];
                    if (bias) v += bias[col];
                    C[coff + (long)row * ldc + col] = v;
                }
            }
        }
    }
}

// ---------------- elementwise / edge kernels ----------------
__device__ __forceinline__ unsigned f2ord(float f) {
    unsigned u = __float_as_uint(f);
    return (u & 0x80000000u) ? ~u : (u | 0x80000000u);
}
__device__ __forceinline__ float ord2f(unsigned u) {
    return (u & 0x80000000u) ? __uint_as_float(u & 0x7FFFFFFFu) : __uint_as_float(~u);
}

__global__ void zero_kernel(float* y, long n) {
    long i = (long)blockIdx.x * blockDim.x + threadIdx.x;
    if (i < n) y[i] = 0.0f;
}

__global__ void gelu_kernel(const float* __restrict__ x, float* __restrict__ y, long n) {
    long i = (long)blockIdx.x * blockDim.x + threadIdx.x;
    if (i < n) {
        float v = x[i];
        y[i] = 0.5f * v * (1.0f + erff(v * 0.70710678118654752f));
    }
}

// y = g*o + (1-g)*x, then leaky_relu(0.01); g = sigmoid(skip[0]); in-place safe
__global__ void combine_kernel(const float* __restrict__ o, const float* __restrict__ x,
                               const float* __restrict__ skip, float* __restrict__ y, long n) {
    long i = (long)blockIdx.x * blockDim.x + threadIdx.x;
    if (i < n) {
        float g = 1.0f / (1.0f + __expf(-skip[0]));
        float v = g * o[i] + (1.0f - g) * x[i];
        y[i] = (v > 0.0f) ? v : 0.01f * v;
    }
}

__global__ void add_kernel(float* __restrict__ y, const float* __restrict__ s, long n) {
    long i = (long)blockIdx.x * blockDim.x + threadIdx.x;
    if (i < n) y[i] += s[i];
}

// alpha[e,h] = (q[dst] . krel[src])_h * p_rel[h] / sqrt(D)
__global__ void edge_alpha_kernel(const float* __restrict__ q, const float* __restrict__ krel,
                                  const int* __restrict__ src, const int* __restrict__ dst,
                                  const float* __restrict__ prel, float* __restrict__ alpha, int E) {
    int eh = blockIdx.x * blockDim.x + threadIdx.x;
    if (eh >= E * NH) return;
    int e = eh >> 3, h = eh & 7;
    int s = src[e], d = dst[e];
    const float4* qv = (const float4*)(q    + (long)d * HID + h * HD);
    const float4* kv = (const float4*)(krel + (long)s * HID + h * HD);
    float acc = 0.0f;
    #pragma unroll
    for (int i = 0; i < HD / 4; ++i) {
        float4 a4 = qv[i], b4 = kv[i];
        acc += a4.x * b4.x + a4.y * b4.y + a4.z * b4.z + a4.w * b4.w;
    }
    alpha[eh] = acc * prel[h] * 0.125f;   // 1/sqrt(64)
}

__global__ void softmax_init_kernel(unsigned* __restrict__ m, float* __restrict__ s, int n) {
    int i = blockIdx.x * blockDim.x + threadIdx.x;
    if (i < n) { m[i] = f2ord(-3.0e38f); s[i] = 0.0f; }
}

__global__ void edge_max_kernel(const float* __restrict__ alpha, const int* __restrict__ dst,
                                unsigned* __restrict__ m, int E) {
    int eh = blockIdx.x * blockDim.x + threadIdx.x;
    if (eh >= E * NH) return;
    int e = eh >> 3, h = eh & 7;
    atomicMax(&m[dst[e] * NH + h], f2ord(alpha[eh]));
}

__global__ void edge_expsum_kernel(float* __restrict__ alpha, const int* __restrict__ dst,
                                   const unsigned* __restrict__ m, float* __restrict__ s, int E) {
    int eh = blockIdx.x * blockDim.x + threadIdx.x;
    if (eh >= E * NH) return;
    int e = eh >> 3, h = eh & 7;
    float mv = ord2f(m[dst[e] * NH + h]);
    float ex = __expf(alpha[eh] - mv);
    alpha[eh] = ex;
    atomicAdd(&s[dst[e] * NH + h], ex);
}

__global__ void edge_msg_kernel(const float* __restrict__ alpha,
                                const int* __restrict__ src, const int* __restrict__ dst,
                                const float* __restrict__ ssum, const float* __restrict__ vrel,
                                float* __restrict__ out, int E) {
    int eh = blockIdx.x * blockDim.x + threadIdx.x;
    if (eh >= E * NH) return;
    int e = eh >> 3, h = eh & 7;
    int s = src[e], d = dst[e];
    float w = alpha[eh] / (ssum[d * NH + h] + 1e-16f);
    const float4* vp = (const float4*)(vrel + (long)s * HID + h * HD);
    float*        op = out + (long)d * HID + h * HD;
    #pragma unroll 4
    for (int i = 0; i < HD / 4; ++i) {
        float4 v4 = vp[i];
        atomicAdd(&op[4 * i + 0], v4.x * w);
        atomicAdd(&op[4 * i + 1], v4.y * w);
        atomicAdd(&op[4 * i + 2], v4.z * w);
        atomicAdd(&op[4 * i + 3], v4.w * w);
    }
}

// out[i] = X[i,:] . w + b[0]   (Kd multiple of 4)
__global__ void gemv_kernel(const float* __restrict__ X, const float* __restrict__ wv,
                            const float* __restrict__ bv, float* __restrict__ out, int M, int Kd) {
    int i = blockIdx.x * blockDim.x + threadIdx.x;
    if (i >= M) return;
    const float4* xp = (const float4*)(X + (long)i * Kd);
    const float4* wp = (const float4*)wv;
    float acc = 0.0f;
    for (int k = 0; k < Kd / 4; ++k) {
        float4 a4 = xp[k], b4 = wp[k];
        acc += a4.x * b4.x + a4.y * b4.y + a4.z * b4.z + a4.w * b4.w;
    }
    out[i] = acc + bv[0];
}

// score = sigmoid( sum_i S[s,i]*T[d,i]*finw[i] + finb )
__global__ void link_kernel(const float* __restrict__ S, const float* __restrict__ T,
                            const int* __restrict__ srcIdx, const int* __restrict__ dstIdx,
                            const float* __restrict__ finw, const float* __restrict__ finb,
                            float* __restrict__ out, int E) {
    int e = blockIdx.x * blockDim.x + threadIdx.x;
    if (e >= E) return;
    const float4* sp = (const float4*)(S + (long)srcIdx[e] * NOUT);
    const float4* tp = (const float4*)(T + (long)dstIdx[e] * NOUT);
    const float4* fp = (const float4*)finw;
    float acc = 0.0f;
    #pragma unroll 4
    for (int i = 0; i < NOUT / 4; ++i) {
        float4 a4 = sp[i], b4 = tp[i], w4 = fp[i];
        acc += a4.x * b4.x * w4.x + a4.y * b4.y * w4.y + a4.z * b4.z * w4.z + a4.w * b4.w * w4.w;
    }
    acc += finb[0];
    out[e] = 1.0f / (1.0f + __expf(-acc));
}

// ---------------- host orchestration ----------------
static inline void launch_gemm(const float* A, int lda, long sA,
                               const float* B, int ldb, long sB,
                               const float* bias,
                               float* C, int ldc, long sC,
                               int M, int Nn, int K, int batch, hipStream_t st) {
    dim3 grid((unsigned)((M + BM - 1) / BM), (unsigned)(Nn / BN), (unsigned)batch);
    wmma_gemm_kernel<<<grid, 256, 0, st>>>(A, lda, sA, B, ldb, sB, bias, C, ldc, sC, M, Nn, K);
}

extern "C" void kernel_launch(void* const* d_in, const int* in_sizes, int n_in,
                              void* d_out, int out_size, void* d_ws, size_t ws_size,
                              hipStream_t stream) {
    const float* x_dev  = (const float*)d_in[0];
    const float* x_repo = (const float*)d_in[1];
    const int*   ei[3]  = { (const int*)d_in[2], (const int*)d_in[3], (const int*)d_in[4] };
    const int*   pos_e  = (const int*)d_in[5];
    const int*   neg_e  = (const int*)d_in[6];
    const float* k_w    = (const float*)d_in[7];
    const float* k_b    = (const float*)d_in[8];
    const float* q_w    = (const float*)d_in[9];
    const float* q_b    = (const float*)d_in[10];
    const float* v_w    = (const float*)d_in[11];
    const float* v_b    = (const float*)d_in[12];
    const float* a_w    = (const float*)d_in[13];
    const float* a_b    = (const float*)d_in[14];
    const float* skip   = (const float*)d_in[15];
    const float* a_rel  = (const float*)d_in[16];
    const float* m_rel  = (const float*)d_in[17];
    const float* p_rel  = (const float*)d_in[18];
    const float* reg_w1 = (const float*)d_in[19];
    const float* reg_b1 = (const float*)d_in[20];
    const float* reg_w2 = (const float*)d_in[21];
    const float* reg_b2 = (const float*)d_in[22];
    const float* src_w  = (const float*)d_in[23];
    const float* src_b  = (const float*)d_in[24];
    const float* dst_w  = (const float*)d_in[25];
    const float* dst_b  = (const float*)d_in[26];
    const float* fin_w  = (const float*)d_in[27];
    const float* fin_b  = (const float*)d_in[28];

    const int Es[3]  = { in_sizes[2] / 2, in_sizes[3] / 2, in_sizes[4] / 2 };
    const int EP     = in_sizes[5] / 2;
    const int stv[3] = { 0, 1, 0 };
    const int dtv[3] = { 1, 0, 0 };
    const int* e_src[3] = { ei[0], ei[1], ei[2] };
    const int* e_dst[3] = { ei[0] + Es[0], ei[1] + Es[1], ei[2] + Es[2] };
    int maxE = Es[0]; if (Es[1] > maxE) maxE = Es[1]; if (Es[2] > maxE) maxE = Es[2];

    // ---- workspace carve ----
    char* wp = (char*)d_ws;
    auto alloc = [&](size_t bytes) -> float* {
        float* p = (float*)wp;
        wp += (bytes + 255) & ~(size_t)255;
        return p;
    };
    const size_t NB = (size_t)NNODE * HID * sizeof(float);
    float* cur[2]  = { alloc(NB), alloc(NB) };
    float* kb2[2]  = { alloc(NB), alloc(NB) };
    float* qb2[2]  = { alloc(NB), alloc(NB) };
    float* vb2[2]  = { alloc(NB), alloc(NB) };
    float* krel    = alloc(NB);
    float* vrel    = alloc(NB);
    float* outb[2] = { alloc(NB), alloc(NB) };
    float* emb[2]  = { alloc(NB), alloc(NB) };
    float* alpha   = alloc((size_t)maxE * NH * sizeof(float));
    unsigned* mbuf = (unsigned*)alloc((size_t)NNODE * NH * sizeof(unsigned));
    float* sbuf    = alloc((size_t)NNODE * NH * sizeof(float));
    // reuse after edge phase of each conv:
    float* tmpG = krel;   // gelu output
    float* obuf = vrel;   // a-linear output
    // reuse at head stage (towers done):
    float* Rbuf = kb2[0]; // [N,256] regression hidden
    float* Sbuf = qb2[0]; // [N,256] src projection
    float* Tbuf = qb2[1]; // [N,256] dst projection

    float* out_f     = (float*)d_out;
    float* fit_dev   = out_f;
    float* fit_repo  = out_f + NNODE;
    float* score_pos = out_f + 2 * NNODE;
    float* score_neg = out_f + 2 * NNODE + EP;

    const long NE   = (long)NNODE * HID;
    const int  EB   = 256;
    const unsigned NEb = (unsigned)((NE + EB - 1) / EB);

    auto conv = [&](int c) {
        // K/Q/V projections for both node types
        for (int t = 0; t < 2; ++t) {
            const size_t wofs = (size_t)(c * 2 + t) * HID * HID;
            const size_t bofs = (size_t)(c * 2 + t) * HID;
            launch_gemm(cur[t], HID, 0, k_w + wofs, HID, 0, k_b + bofs, kb2[t], HID, 0,
                        NNODE, HID, HID, 1, stream);
            launch_gemm(cur[t], HID, 0, q_w + wofs, HID, 0, q_b + bofs, qb2[t], HID, 0,
                        NNODE, HID, HID, 1, stream);
            launch_gemm(cur[t], HID, 0, v_w + wofs, HID, 0, v_b + bofs, vb2[t], HID, 0,
                        NNODE, HID, HID, 1, stream);
        }
        zero_kernel<<<NEb, EB, 0, stream>>>(outb[0], NE);
        zero_kernel<<<NEb, EB, 0, stream>>>(outb[1], NE);

        for (int r = 0; r < 3; ++r) {
            const int st = stv[r], dt = dtv[r], E = Es[r];
            const size_t relofs = (size_t)(c * 3 + r) * NH * HD * HD;
            // per-head relation einsums as batched WMMA GEMMs (batch = 8 heads)
            launch_gemm(kb2[st], HID, HD, a_rel + relofs, HD, (long)HD * HD, nullptr,
                        krel, HID, HD, NNODE, HD, HD, NH, stream);
            launch_gemm(vb2[st], HID, HD, m_rel + relofs, HD, (long)HD * HD, nullptr,
                        vrel, HID, HD, NNODE, HD, HD, NH, stream);
            // edge attention + segment softmax + message scatter
            const unsigned EHb = (unsigned)(((long)E * NH + EB - 1) / EB);
            edge_alpha_kernel<<<EHb, EB, 0, stream>>>(qb2[dt], krel, e_src[r], e_dst[r],
                                                      p_rel + (size_t)(c * 3 + r) * NH, alpha, E);
            softmax_init_kernel<<<(NNODE * NH + EB - 1) / EB, EB, 0, stream>>>(mbuf, sbuf, NNODE * NH);
            edge_max_kernel<<<EHb, EB, 0, stream>>>(alpha, e_dst[r], mbuf, E);
            edge_expsum_kernel<<<EHb, EB, 0, stream>>>(alpha, e_dst[r], mbuf, sbuf, E);
            edge_msg_kernel<<<EHb, EB, 0, stream>>>(alpha, e_src[r], e_dst[r], sbuf, vrel, outb[dt], E);
        }

        // gelu -> A-linear -> skip gate (+leaky)
        for (int t = 0; t < 2; ++t) {
            gelu_kernel<<<NEb, EB, 0, stream>>>(outb[t], tmpG, NE);
            launch_gemm(tmpG, HID, 0, a_w + (size_t)(c * 2 + t) * HID * HID, HID, 0,
                        a_b + (size_t)(c * 2 + t) * HID, obuf, HID, 0, NNODE, HID, HID, 1, stream);
            combine_kernel<<<NEb, EB, 0, stream>>>(obuf, cur[t], skip + (c * 2 + t), cur[t], NE);
        }
    };

    // ---- tower 1 (convs 0,1) ----
    hipMemcpyAsync(cur[0], x_dev, NB, hipMemcpyDeviceToDevice, stream);
    hipMemcpyAsync(cur[1], x_repo, NB, hipMemcpyDeviceToDevice, stream);
    conv(0); conv(1);
    hipMemcpyAsync(emb[0], cur[0], NB, hipMemcpyDeviceToDevice, stream);
    hipMemcpyAsync(emb[1], cur[1], NB, hipMemcpyDeviceToDevice, stream);

    // ---- tower 2 (convs 2,3) ----
    hipMemcpyAsync(cur[0], x_dev, NB, hipMemcpyDeviceToDevice, stream);
    hipMemcpyAsync(cur[1], x_repo, NB, hipMemcpyDeviceToDevice, stream);
    conv(2); conv(3);
    add_kernel<<<NEb, EB, 0, stream>>>(emb[0], cur[0], NE);
    add_kernel<<<NEb, EB, 0, stream>>>(emb[1], cur[1], NE);

    // ---- fitness heads ----
    launch_gemm(emb[0], HID, 0, reg_w1, NOUT, 0, reg_b1, Rbuf, NOUT, 0, NNODE, NOUT, HID, 1, stream);
    gemv_kernel<<<(NNODE + EB - 1) / EB, EB, 0, stream>>>(Rbuf, reg_w2, reg_b2, fit_dev, NNODE, NOUT);
    launch_gemm(emb[1], HID, 0, reg_w1, NOUT, 0, reg_b1, Rbuf, NOUT, 0, NNODE, NOUT, HID, 1, stream);
    gemv_kernel<<<(NNODE + EB - 1) / EB, EB, 0, stream>>>(Rbuf, reg_w2, reg_b2, fit_repo, NNODE, NOUT);

    // ---- link heads: project all nodes once, then per-edge dot ----
    launch_gemm(emb[0], HID, 0, src_w, NOUT, 0, src_b, Sbuf, NOUT, 0, NNODE, NOUT, HID, 1, stream);
    launch_gemm(emb[1], HID, 0, dst_w, NOUT, 0, dst_b, Tbuf, NOUT, 0, NNODE, NOUT, HID, 1, stream);
    link_kernel<<<(EP + EB - 1) / EB, EB, 0, stream>>>(Sbuf, Tbuf, pos_e, pos_e + EP,
                                                       fin_w, fin_b, score_pos, EP);
    link_kernel<<<(EP + EB - 1) / EB, EB, 0, stream>>>(Sbuf, Tbuf, neg_e, neg_e + EP,
                                                       fin_w, fin_b, score_neg, EP);
    (void)n_in; (void)out_size; (void)ws_size;
}
